// OneBodyInteractionEnergyReadout_13872744366323
// MI455X (gfx1250) — matrix operand order repaired and checked by reading, not compiled, not run on hardware
//
#include <hip/hip_runtime.h>
#include <hip/hip_bf16.h>

#define N_NODES   50000
#define N_EDGES   1600000
#define Q_MULT    8
#define NODE_MULT 64
#define RBF       8
#define HID       64

typedef __attribute__((ext_vector_type(16))) _Float16 v16h;
typedef __attribute__((ext_vector_type(8)))  float    v8f;

#define TILE_E          16
#define WAVES_PER_BLOCK 8
#define EDGES_PER_BLOCK (TILE_E * WAVES_PER_BLOCK)

// Fast SiLU: single v_rcp_f32 instead of IEEE divide sequence.
__device__ __forceinline__ float silu_f(float x) {
    return x * __builtin_amdgcn_rcpf(1.0f + __expf(-x));
}

// K-permutation making A-fragment reads contiguous per lane.
// For lane group g (lanes 0-15 -> g=0, 16-31 -> g=1) and K-chunk kb, the A frag
// needs k = kb + {0..7, 16..23} (g=0) or kb + {8..15, 24..31} (g=1) in j-order.
// sigma maps those sets to 16 consecutive slots.
__device__ __forceinline__ int sigma64(int k) {
    return ((k >> 5) << 5) | (((k >> 3) & 1) << 4) | (((k >> 4) & 1) << 3) | (k & 7);
}

__global__ void zero_ws_kernel(float* __restrict__ p, int n) {
    int i = blockIdx.x * blockDim.x + threadIdx.x;
    if (i < n) p[i] = 0.0f;
}

// ---------------------------------------------------------------------------
// Edge MLP (WMMA f16 -> f32) + gather(charges_induced[sender]) + scatter-add
// into sum0 (N x 8) / sum1 (N x 24) by receiver.
//
// LDS layouts (all fragment loads are contiguous 32B per lane -> ds_load_b128x2):
//  - weights: per n-tile, per 16-wide k-group, a 16x16 block stored [n&15][k&15]
//  - h tiles: h[m][k], plain row-major
// ---------------------------------------------------------------------------
__launch_bounds__(256, 1)
__global__ void edge_mlp_scatter_kernel(const float* __restrict__ edge_feats,   // E x 8
                                        const float* __restrict__ edge_attrs,   // E x 4
                                        const float* __restrict__ charges_ind,  // N x 8
                                        const float* __restrict__ w1,           // 8 x 64
                                        const float* __restrict__ w2,           // 64 x 64
                                        const float* __restrict__ w3,           // 64 x 64
                                        const float* __restrict__ w4,           // 64 x 16
                                        const int*   __restrict__ edge_index,   // 2 x E
                                        float* __restrict__ sum0,               // N x 8
                                        float* __restrict__ sum1)               // N x 24
{
    __shared__ alignas(32) _Float16 lds_w1[32 * 64];   // K padded 8 -> 32 (zeros)
    __shared__ alignas(32) _Float16 lds_w2[64 * 64];
    __shared__ alignas(32) _Float16 lds_w3[64 * 64];
    __shared__ alignas(32) _Float16 lds_w4[64 * 16];
    __shared__ alignas(32) _Float16 lds_h[WAVES_PER_BLOCK][2][16 * 64];

    const int tid = threadIdx.x;

    // ---- stage weights into block-fragment layout, folding 1/sqrt(K), f16 ----
    const float s8  = 0.35355339059327373f;  // 1/sqrt(8)
    const float s64 = 0.125f;                // 1/sqrt(64)
    for (int i = tid; i < 32 * 64; i += 256) {            // w1: K=32 (padded), N=64
        int n = i >> 5, k = i & 31;
        int addr = (n >> 4) * 512 + (k >> 4) * 256 + (n & 15) * 16 + (k & 15);
        lds_w1[addr] = (_Float16)(k < RBF ? w1[k * 64 + n] * s8 : 0.0f);
    }
    for (int i = tid; i < 64 * 64; i += 256) {            // w2: K=64, N=64
        int n = i >> 6, k = i & 63;
        int addr = (n >> 4) * 1024 + (k >> 4) * 256 + (n & 15) * 16 + (k & 15);
        lds_w2[addr] = (_Float16)(w2[k * 64 + n] * s64);
    }
    for (int i = tid; i < 64 * 64; i += 256) {            // w3: K=64, N=64
        int n = i >> 6, k = i & 63;
        int addr = (n >> 4) * 1024 + (k >> 4) * 256 + (n & 15) * 16 + (k & 15);
        lds_w3[addr] = (_Float16)(w3[k * 64 + n] * s64);
    }
    for (int i = tid; i < 64 * 16; i += 256) {            // w4: K=64, N=16
        int n = i >> 6, k = i & 63;
        int addr = (k >> 4) * 256 + n * 16 + (k & 15);
        lds_w4[addr] = (_Float16)(w4[k * 16 + n] * s64);
    }
    __syncthreads();

    const int wave = tid >> 5;
    const int lane = tid & 31;
    const int half = lane >> 4;   // lane group 0/1
    const int mn   = lane & 15;   // A-row (M) / B,D column (N) owned by this lane
    const int ebase = blockIdx.x * EDGES_PER_BLOCK + wave * TILE_E;

    _Float16* h_cur = lds_h[wave][0];
    _Float16* h_nxt = lds_h[wave][1];

    // ---------------- Layer 1: edge_feats (16x8, K padded to 32) @ W1 -> 16x64
    {
        v16h a;
        #pragma unroll
        for (int j = 0; j < 16; ++j) a[j] = (_Float16)0.0f;
        int e = ebase + mn;
        if (half == 0 && e < N_EDGES) {
            const float* ef = edge_feats + (size_t)e * RBF;
            #pragma unroll
            for (int j = 0; j < RBF; ++j) a[j] = (_Float16)ef[j];  // 2x b128 loads
        }
        #pragma unroll
        for (int nt = 0; nt < 4; ++nt) {
            v16h b = *(const v16h*)&lds_w1[nt * 512 + half * 256 + mn * 16];
            v8f cz = {};
            v8f d  = __builtin_amdgcn_wmma_f32_16x16x32_f16(false, a, false, b,
                                                            (short)0, cz, false, false);
            int sc = nt * 16 + mn;
            #pragma unroll
            for (int r = 0; r < 8; ++r)
                h_cur[(r + 8 * half) * 64 + sc] = (_Float16)silu_f(d[r]);
        }
    }
    __syncthreads();

    // ---------------- Layers 2 & 3: 16x64 @ 64x64, two K=32 chunks
    #pragma unroll
    for (int layer = 0; layer < 2; ++layer) {
        const _Float16* w = layer ? lds_w3 : lds_w2;
        v16h a0 = *(const v16h*)&h_cur[mn * 64 + (half << 4)];        // kb = 0
        v16h a1 = *(const v16h*)&h_cur[mn * 64 + 32 + (half << 4)];   // kb = 32
        #pragma unroll
        for (int nt = 0; nt < 4; ++nt) {
            v16h b0 = *(const v16h*)&w[nt * 1024 + half * 256 + mn * 16];
            v16h b1 = *(const v16h*)&w[nt * 1024 + (2 + half) * 256 + mn * 16];
            v8f acc = {};
            acc = __builtin_amdgcn_wmma_f32_16x16x32_f16(false, a0, false, b0,
                                                         (short)0, acc, false, false);
            acc = __builtin_amdgcn_wmma_f32_16x16x32_f16(false, a1, false, b1,
                                                         (short)0, acc, false, false);
            int sc = nt * 16 + mn;
            #pragma unroll
            for (int r = 0; r < 8; ++r)
                h_nxt[(r + 8 * half) * 64 + sc] = (_Float16)silu_f(acc[r]);
        }
        _Float16* t = h_cur; h_cur = h_nxt; h_nxt = t;
        __syncthreads();
    }

    // ---------------- Layer 4: 16x64 @ 64x16 -> tp_w tile 16x16 (f32)
    float* dtile = (float*)h_nxt;   // 16x16 f32 = 1 KB inside the free 2 KB buffer
    {
        v16h a0 = *(const v16h*)&h_cur[mn * 64 + (half << 4)];
        v16h a1 = *(const v16h*)&h_cur[mn * 64 + 32 + (half << 4)];
        v16h b0 = *(const v16h*)&lds_w4[half * 256 + mn * 16];
        v16h b1 = *(const v16h*)&lds_w4[(2 + half) * 256 + mn * 16];
        v8f acc = {};
        acc = __builtin_amdgcn_wmma_f32_16x16x32_f16(false, a0, false, b0,
                                                     (short)0, acc, false, false);
        acc = __builtin_amdgcn_wmma_f32_16x16x32_f16(false, a1, false, b1,
                                                     (short)0, acc, false, false);
        #pragma unroll
        for (int r = 0; r < 8; ++r)
            dtile[(r + 8 * half) * 16 + mn] = acc[r];
    }
    __syncthreads();

    // ---------------- Scatter: 32 output components per edge, one per lane.
    for (int e = 0; e < TILE_E; ++e) {
        int eg = ebase + e;
        if (eg >= N_EDGES) break;
        int snd = edge_index[eg];
        int rcv = edge_index[N_EDGES + eg];
        if (lane < 8) {
            int u = lane;
            float v = dtile[e * 16 + u]
                    * charges_ind[snd * Q_MULT + u]
                    * edge_attrs[eg * 4 + 0];
            unsafeAtomicAdd(&sum0[rcv * Q_MULT + u], v);
        } else {
            int idx = lane - 8;           // 0..23
            int u = idx / 3, kk = idx % 3;
            float v = dtile[e * 16 + 8 + u]
                    * charges_ind[snd * Q_MULT + u]
                    * edge_attrs[eg * 4 + 1 + kk];
            unsafeAtomicAdd(&sum1[rcv * (Q_MULT * 3) + u * 3 + kk], v);
        }
    }
}

// ---------------------------------------------------------------------------
// Per-node readout: out[n] = c * (sum0.(W0.node0) + sum1.(W1.node1)/sqrt(3))
// One wave per node, shfl_xor reduction.
// ---------------------------------------------------------------------------
__launch_bounds__(256, 1)
__global__ void readout_kernel(const float* __restrict__ node_feats,  // N x 256
                               const float* __restrict__ sum0,        // N x 8
                               const float* __restrict__ sum1,        // N x 24
                               const float* __restrict__ W0,          // 8 x 64
                               const float* __restrict__ W1,          // 8 x 64
                               float* __restrict__ out)               // N
{
    __shared__ float sW0[8 * 64];
    __shared__ float sW1[8 * 64];
    for (int i = threadIdx.x; i < 8 * 64; i += 256) { sW0[i] = W0[i]; sW1[i] = W1[i]; }
    __syncthreads();

    const int wave = threadIdx.x >> 5;
    const int lane = threadIdx.x & 31;
    const int n = blockIdx.x * WAVES_PER_BLOCK + wave;
    if (n >= N_NODES) return;

    float s0[8];
    #pragma unroll
    for (int u = 0; u < 8; ++u) s0[u] = sum0[n * 8 + u];
    float s1[8][3];
    #pragma unroll
    for (int u = 0; u < 8; ++u)
        #pragma unroll
        for (int k = 0; k < 3; ++k) s1[u][k] = sum1[n * 24 + u * 3 + k];

    const float inv_sqrt3 = 0.5773502691896258f;
    float acc = 0.0f;
    #pragma unroll
    for (int vv = 0; vv < 2; ++vv) {
        int v = lane + vv * 32;
        float t0 = 0.0f;
        #pragma unroll
        for (int u = 0; u < 8; ++u) t0 += s0[u] * sW0[u * 64 + v];
        acc += node_feats[n * 256 + v] * t0;

        float t1[3] = {0.0f, 0.0f, 0.0f};
        #pragma unroll
        for (int u = 0; u < 8; ++u) {
            float w = sW1[u * 64 + v];
            #pragma unroll
            for (int k = 0; k < 3; ++k) t1[k] += s1[u][k] * w;
        }
        float p1 = 0.0f;
        #pragma unroll
        for (int k = 0; k < 3; ++k)
            p1 += node_feats[n * 256 + NODE_MULT + v * 3 + k] * t1[k];
        acc += p1 * inv_sqrt3;
    }

    #pragma unroll
    for (int off = 16; off > 0; off >>= 1) acc += __shfl_xor(acc, off, 32);
    if (lane == 0) out[n] = acc * (1.0f / 32.0f);   // c = 1/sqrt(8*64*2)
}

extern "C" void kernel_launch(void* const* d_in, const int* in_sizes, int n_in,
                              void* d_out, int out_size, void* d_ws, size_t ws_size,
                              hipStream_t stream) {
    const float* node_feats  = (const float*)d_in[0];
    const float* charges_ind = (const float*)d_in[2];
    const float* edge_feats  = (const float*)d_in[3];
    const float* edge_attrs  = (const float*)d_in[4];
    const float* mlp_w1      = (const float*)d_in[6];
    const float* mlp_w2      = (const float*)d_in[7];
    const float* mlp_w3      = (const float*)d_in[8];
    const float* mlp_w4      = (const float*)d_in[9];
    const float* W0          = (const float*)d_in[10];
    const float* W1          = (const float*)d_in[11];
    const int*   edge_index  = (const int*)d_in[12];
    float*       out         = (float*)d_out;

    float* sum0 = (float*)d_ws;                      // N_NODES * 8
    float* sum1 = sum0 + (size_t)N_NODES * Q_MULT;   // N_NODES * 24

    {
        int n = N_NODES * (Q_MULT + Q_MULT * 3);
        int blocks = (n + 255) / 256;
        zero_ws_kernel<<<blocks, 256, 0, stream>>>(sum0, n);
    }
    {
        int blocks = (N_EDGES + EDGES_PER_BLOCK - 1) / EDGES_PER_BLOCK;
        edge_mlp_scatter_kernel<<<blocks, 256, 0, stream>>>(
            edge_feats, edge_attrs, charges_ind,
            mlp_w1, mlp_w2, mlp_w3, mlp_w4,
            edge_index, sum0, sum1);
    }
    {
        int blocks = (N_NODES + WAVES_PER_BLOCK - 1) / WAVES_PER_BLOCK;
        readout_kernel<<<blocks, 256, 0, stream>>>(
            node_feats, sum0, sum1, W0, W1, out);
    }
}
